// messagePassingConv_5995774345718
// MI455X (gfx1250) — matrix operand order repaired and checked by reading, not compiled, not run on hardware
//
#include <hip/hip_runtime.h>

#define F 128
#define KP 136              // padded LDS row pitch (elements): 272B -> distinct banks, 16B aligned
#define BN_EPS 1e-3f

typedef __attribute__((ext_vector_type(8)))  float   v8f;
typedef __attribute__((ext_vector_type(8)))  __bf16  v8bf;
typedef __attribute__((ext_vector_type(16))) __bf16  v16bf;

__device__ __forceinline__ v16bf vcombine(v8bf lo, v8bf hi) {
  return __builtin_shufflevector(lo, hi, 0,1,2,3,4,5,6,7,8,9,10,11,12,13,14,15);
}

// A-fragment (16x32 bf16 layout): lane l -> row m=l&15, khalf=l>>4.
// K runs: [kb .. kb+7] and [kb+16 .. kb+23], kb = t*32 + khalf*8.
__device__ __forceinline__ v16bf afrag_sum_f32(const float* __restrict__ ra,
                                               const float* __restrict__ rb, int kb) {
  v8f a0 = *(const v8f*)(ra + kb);
  v8f a1 = *(const v8f*)(ra + kb + 16);
  v8f b0 = *(const v8f*)(rb + kb);
  v8f b1 = *(const v8f*)(rb + kb + 16);
  v8bf lo = __builtin_convertvector(a0 + b0, v8bf);
  v8bf hi = __builtin_convertvector(a1 + b1, v8bf);
  return vcombine(lo, hi);
}

__device__ __forceinline__ v16bf afrag_f32(const float* __restrict__ r, int kb) {
  v8f a0 = *(const v8f*)(r + kb);
  v8f a1 = *(const v8f*)(r + kb + 16);
  return vcombine(__builtin_convertvector(a0, v8bf), __builtin_convertvector(a1, v8bf));
}

__device__ __forceinline__ v16bf afrag_lds(const __bf16* p, int kb) {
  v8bf lo = *(const v8bf*)(p + kb);
  v8bf hi = *(const v8bf*)(p + kb + 16);
  return vcombine(lo, hi);
}

// B-fragment (32x16 bf16): lane l -> col n, khalf=l>>4 holds K = khalf*8 + [0..7] and 16 + khalf*8 + [0..7].
// Weights stored transposed in LDS: Wt[n*KP + k].
__device__ __forceinline__ v16bf bfrag_lds(const __bf16* Wt, int n, int t, int khalf) {
  const __bf16* p = Wt + n * KP + t * 32 + khalf * 8;
  v8bf lo = *(const v8bf*)(p);
  v8bf hi = *(const v8bf*)(p + 16);
  return vcombine(lo, hi);
}

__device__ __forceinline__ v8f wmma_bf16(v16bf a, v16bf b, v8f c) {
  return __builtin_amdgcn_wmma_f32_16x16x32_bf16(false, a, false, b, (short)0, c, false, false);
}

__device__ __forceinline__ float sigmoidf_(float t) { return 1.0f / (1.0f + __expf(-t)); }

__global__ __launch_bounds__(256, 1) void fused_msgpass_gru(
    const float* __restrict__ x,
    const int*   __restrict__ pprev,
    const int*   __restrict__ pnext,
    const float* __restrict__ wNext,
    const float* __restrict__ wPrev,
    const float* __restrict__ bias,
    const float* __restrict__ gk,
    const float* __restrict__ gr,
    const float* __restrict__ gbias,
    const float* __restrict__ bn_g,
    const float* __restrict__ bn_b,
    const float* __restrict__ bn_m,
    const float* __restrict__ bn_v,
    float* __restrict__ out,
    int N, int nChunks)
{
  extern __shared__ __align__(32) char smem[];
  __bf16* wNextT = (__bf16*)smem;                 // 128 x KP
  __bf16* wPrevT = wNextT + F * KP;               // 128 x KP
  __bf16* gkT    = wPrevT + F * KP;               // 384 x KP
  __bf16* grT    = gkT + 3 * F * KP;              // 384 x KP
  __bf16* aT     = grT + 3 * F * KP;              // 128 x KP (per-wave 16-row slices)

  const int tid = threadIdx.x;

  // Stage all weights into LDS: f32 -> bf16, transposed Wt[n][k]
  for (int i = tid; i < F * F; i += 256) {
    int k = i >> 7, n = i & (F - 1);
    wNextT[n * KP + k] = (__bf16)wNext[i];
    wPrevT[n * KP + k] = (__bf16)wPrev[i];
  }
  for (int i = tid; i < F * 3 * F; i += 256) {
    int k = i / (3 * F), n = i - k * (3 * F);
    gkT[n * KP + k] = (__bf16)gk[i];
    grT[n * KP + k] = (__bf16)gr[i];
  }
  __syncthreads();

  const int wv    = tid >> 5;    // wave id (8 waves, 16 rows each)
  const int lane  = tid & 31;
  const int nl    = lane & 15;
  const int khalf = lane >> 4;
  __bf16* aRow = aT + (wv * 16 + nl) * KP;

  for (int chunk = blockIdx.x; chunk < nChunks; chunk += (int)gridDim.x) {
    const int rbase = chunk * 128 + wv * 16;
    const int rA  = min(rbase + nl, N - 1);   // row this lane feeds A-fragments for
    // fused gather: pairs[:,0] is repeat(arange(N),2) -> node r uses edges 2r, 2r+1
    const int jp0 = pprev[4 * rA + 1], jp1 = pprev[4 * rA + 3];
    const int jn0 = pnext[4 * rA + 1], jn1 = pnext[4 * rA + 3];
    const float* xp0 = x + (size_t)jp0 * F;
    const float* xp1 = x + (size_t)jp1 * F;
    const float* xn0 = x + (size_t)jn0 * F;
    const float* xn1 = x + (size_t)jn1 * F;
    const float* xs  = x + (size_t)rA  * F;

    v16bf pA[4], nA[4], xA[4];
    #pragma unroll
    for (int t = 0; t < 4; ++t) {
      const int kb = t * 32 + khalf * 8;
      pA[t] = afrag_sum_f32(xp0, xp1, kb);   // prev segment-sum, bf16
      nA[t] = afrag_sum_f32(xn0, xn1, kb);   // next segment-sum, bf16
      xA[t] = afrag_f32(xs, kb);             // GRU state h = x, bf16
    }

    // ---- stage 1: aggre = next@Wn + prev@Wp + b + x ; a = BN(relu(aggre)) ----
    for (int i = 0; i < 8; ++i) {
      const int col = i * 16 + nl;
      v8f acc = {};
      #pragma unroll
      for (int t = 0; t < 4; ++t) {
        acc = wmma_bf16(nA[t], bfrag_lds(wNextT, col, t, khalf), acc);
        acc = wmma_bf16(pA[t], bfrag_lds(wPrevT, col, t, khalf), acc);
      }
      const float bb = bias[col];
      const float g  = bn_g[col], be = bn_b[col], mu = bn_m[col];
      const float iv = rsqrtf(bn_v[col] + BN_EPS);
      #pragma unroll
      for (int v = 0; v < 8; ++v) {
        const int rr  = rbase + v + 8 * khalf;   // C/D layout: lanes16-31 hold rows 8..15
        const int rrc = min(rr, N - 1);
        const float xi = x[(size_t)rrc * F + col];
        float f = acc[v] + bb + xi;
        f = fmaxf(f, 0.0f);
        f = g * (f - mu) * iv + be;
        aT[(wv * 16 + v + 8 * khalf) * KP + col] = (__bf16)f;
      }
    }

    // ---- stage 2: GRU (reset_after): mx = a@gk + b0 ; mh = x@gr + b1 ----
    v16bf aA[4];
    #pragma unroll
    for (int t = 0; t < 4; ++t)
      aA[t] = afrag_lds(aRow, t * 32 + khalf * 8);   // same-wave LDS RAW, hw-ordered

    for (int j = 0; j < 8; ++j) {
      const int col = j * 16 + nl;
      v8f axz = {}, ahz = {}, axr = {}, ahr = {}, axh = {}, ahh = {};
      #pragma unroll
      for (int t = 0; t < 4; ++t) {
        axz = wmma_bf16(aA[t], bfrag_lds(gkT, col,         t, khalf), axz);
        ahz = wmma_bf16(xA[t], bfrag_lds(grT, col,         t, khalf), ahz);
        axr = wmma_bf16(aA[t], bfrag_lds(gkT, col + F,     t, khalf), axr);
        ahr = wmma_bf16(xA[t], bfrag_lds(grT, col + F,     t, khalf), ahr);
        axh = wmma_bf16(aA[t], bfrag_lds(gkT, col + 2 * F, t, khalf), axh);
        ahh = wmma_bf16(xA[t], bfrag_lds(grT, col + 2 * F, t, khalf), ahh);
      }
      const float bz0 = gbias[col],         bz1 = gbias[3 * F + col];
      const float br0 = gbias[F + col],     br1 = gbias[4 * F + col];
      const float bh0 = gbias[2 * F + col], bh1 = gbias[5 * F + col];
      #pragma unroll
      for (int v = 0; v < 8; ++v) {
        const int rr  = rbase + v + 8 * khalf;
        const int rrc = min(rr, N - 1);
        const float xi = x[(size_t)rrc * F + col];
        const float z  = sigmoidf_((axz[v] + bz0) + (ahz[v] + bz1));
        const float rg = sigmoidf_((axr[v] + br0) + (ahr[v] + br1));
        const float hh = tanhf((axh[v] + bh0) + rg * (ahh[v] + bh1));
        const float o  = z * xi + (1.0f - z) * hh;
        if (rr < N) out[(size_t)rr * F + col] = o;
      }
    }
  }
}

extern "C" void kernel_launch(void* const* d_in, const int* in_sizes, int n_in,
                              void* d_out, int out_size, void* d_ws, size_t ws_size,
                              hipStream_t stream) {
  const float* x   = (const float*)d_in[0];
  const int* pprev = (const int*)d_in[1];
  const int* pnext = (const int*)d_in[2];
  const float* wN  = (const float*)d_in[3];
  const float* wP  = (const float*)d_in[4];
  const float* b   = (const float*)d_in[5];
  const float* gk  = (const float*)d_in[6];
  const float* gr  = (const float*)d_in[7];
  const float* gb  = (const float*)d_in[8];
  const float* bng = (const float*)d_in[9];
  const float* bnb = (const float*)d_in[10];
  const float* bnm = (const float*)d_in[11];
  const float* bnv = (const float*)d_in[12];
  float* out = (float*)d_out;

  const int N = in_sizes[0] / F;
  const int nChunks = (N + 127) / 128;

  constexpr size_t SMEM_BYTES =
      (size_t)(3 * F * KP + 2 * 3 * F * KP) * sizeof(unsigned short);  // 313,344 B < 320 KB

  (void)hipFuncSetAttribute((const void*)fused_msgpass_gru,
                            hipFuncAttributeMaxDynamicSharedMemorySize, (int)SMEM_BYTES);

  int grid = nChunks < 1024 ? nChunks : 1024;   // persistent blocks amortize weight staging
  fused_msgpass_gru<<<grid, 256, SMEM_BYTES, stream>>>(
      x, pprev, pnext, wN, wP, b, gk, gr, gb, bng, bnb, bnm, bnv, out, N, nChunks);
}
